// MambaBackend_74105365725306
// MI455X (gfx1250) — matrix operand order repaired and checked by reading, not compiled, not run on hardware
//
#include <hip/hip_runtime.h>
#include <hip/hip_bf16.h>

// ---------------- problem constants (match reference) ----------------
#define D_MODEL_C 256
#define N_LAYERS_C 6
#define D_STATE_C 16
#define D_CONV_C 4
#define D_INNER_C 512          // EXPAND * D_MODEL
#define DT_RANK_C 16
#define BATCH_C 8
#define SEQ_C 4096
#define BL_C (BATCH_C * SEQ_C) // 32768 tokens

// padded GEMM dims so every tile is exact (no guards -> pure b128 traffic)
#define XDB_N 128              // W_x rows 48 -> 128
#define DT_K 32                // W_dt K 16 -> 32

typedef __attribute__((ext_vector_type(16))) __bf16 v16bf;
typedef __attribute__((ext_vector_type(8)))  __bf16 v8bf;
typedef __attribute__((ext_vector_type(8)))  float  v8f;
typedef __attribute__((ext_vector_type(4)))  int    v4i;

#if __has_builtin(__builtin_amdgcn_global_load_async_to_lds_b128)
#define HAVE_ASYNC_LDS 1
typedef __attribute__((address_space(1))) v4i gv4i;   // global int4
typedef __attribute__((address_space(3))) v4i lv4i;   // LDS int4
#else
#define HAVE_ASYNC_LDS 0
#endif

static __device__ __forceinline__ __bf16 f2bf(float f) {
  unsigned int u = __builtin_bit_cast(unsigned int, f);
  unsigned int r = (u + 0x7FFFu + ((u >> 16) & 1u)) >> 16;   // RNE
  unsigned short s = (unsigned short)r;
  return __builtin_bit_cast(__bf16, s);
}
static __device__ __forceinline__ __bf16 bf_zero() {
  unsigned short s = 0;
  return __builtin_bit_cast(__bf16, s);
}

// ------- fp32 [L][Nin][Kin] -> bf16 [L][Nout][Kout], zero padded (grid-stride) -------
__global__ void k_cvt_pad(const float* __restrict__ in, __bf16* __restrict__ out,
                          int L, int Nin, int Nout, int Kin, int Kout) {
  const long total = (long)L * Nout * Kout;
  long i = (long)blockIdx.x * blockDim.x + threadIdx.x;
  const long stride = (long)gridDim.x * blockDim.x;
  for (; i < total; i += stride) {
    const int k = (int)(i % Kout);
    const long t = i / Kout;
    const int n = (int)(t % Nout);
    const int l = (int)(t / Nout);
    out[i] = (n < Nin && k < Kin) ? f2bf(in[((size_t)l * Nin + n) * Kin + k]) : bf_zero();
  }
}

// ---------------- tiled transpose: in[b] is [R][C] -> out[b] is [C][R] ----------------
__global__ void k_transpose(const float* __restrict__ in, float* __restrict__ out, int R, int C) {
  __shared__ float tile[32][33];
  const int b = blockIdx.z;
  const float* ip = in + (size_t)b * R * C;
  float* op = out + (size_t)b * R * C;
  const int c0 = blockIdx.x * 32;
  const int r0 = blockIdx.y * 32;
  for (int yy = threadIdx.y; yy < 32; yy += 8)
    tile[yy][threadIdx.x] = ip[(size_t)(r0 + yy) * C + c0 + threadIdx.x];
  __syncthreads();
  for (int yy = threadIdx.y; yy < 32; yy += 8)
    op[(size_t)(c0 + yy) * R + r0 + threadIdx.x] = tile[threadIdx.x][yy];
}

// ---------------- LayerNorm over D_MODEL, output bf16 ----------------
__global__ void k_layernorm(const float* __restrict__ h, const float* __restrict__ g,
                            const float* __restrict__ b, __bf16* __restrict__ o) {
  __shared__ float red[D_MODEL_C];
  const int t = blockIdx.x;
  const int i = threadIdx.x;
  const float v = h[(size_t)t * D_MODEL_C + i];
  red[i] = v;
  __syncthreads();
  for (int s = D_MODEL_C / 2; s > 0; s >>= 1) {
    if (i < s) red[i] += red[i + s];
    __syncthreads();
  }
  const float mean = red[0] * (1.0f / D_MODEL_C);
  __syncthreads();
  const float dv = v - mean;
  red[i] = dv * dv;
  __syncthreads();
  for (int s = D_MODEL_C / 2; s > 0; s >>= 1) {
    if (i < s) red[i] += red[i + s];
    __syncthreads();
  }
  const float var = red[0] * (1.0f / D_MODEL_C);
  o[(size_t)t * D_MODEL_C + i] = f2bf(dv * rsqrtf(var + 1e-5f) * g[i] + b[i]);
}

// ---------------- WMMA bf16 GEMM: C[M,N] = A[M,K] * W[N,K]^T ----------------
// Requirements (enforced by padding): M % 64 == 0, N % 128 == 0, K % 32 == 0.
// Block tile 64(M) x 128(N), 256 threads = 8 wave32 waves arranged 2x4,
// each wave computes 2x2 fragments of 16x16 via v_wmma_f32_16x16x32_bf16.
// Tile staging uses GLOBAL_LOAD_ASYNC_TO_LDS_B128 (ASYNCcnt) when available,
// else 16B global load + 16B LDS store. LDS row stride 40 bf16 (80B):
// 16B-aligned AND 20-bank row stride (conflict-free fragment gathers).
// MODE: 0 = plain, 1 = +bias then softplus, 2 = +res (residual add)
#define LDSTR 40
template <int MODE>
__global__ __launch_bounds__(256) void k_gemm_bf16_wmma(
    const __bf16* __restrict__ A, const __bf16* __restrict__ W,
    float* __restrict__ C, int M, int N, int K,
    const float* __restrict__ bias, const float* __restrict__ res) {
  __shared__ __align__(16) __bf16 As[64][LDSTR];
  __shared__ __align__(16) __bf16 Bs[128][LDSTR];

  const int tid  = threadIdx.x;
  const int lane = tid & 31;
  const int wave = tid >> 5;
  const int wm = wave >> 2;             // 0..1  (32-row group)
  const int wn = wave & 3;              // 0..3  (32-col group)
  const int m0 = blockIdx.y * 64;
  const int n0 = blockIdx.x * 128;

  const int arow = tid >> 2, acol = (tid & 3) * 8;     // A: 64 x 32, 8 elems/thread
  const int brow = tid >> 1, bcol = (tid & 1) * 16;    // B: 128 x 32, 16 elems/thread
  const __bf16* aptr = A + (size_t)(m0 + arow) * K + acol;
  const __bf16* bptr = W + (size_t)(n0 + brow) * K + bcol;

  v8f acc[2][2] = {};

  for (int k0 = 0; k0 < K; k0 += 32) {
    if (k0 + 32 < K) {   // speculative prefetch of the K-slab at k0 = 32 (address does not advance with k0)
      __builtin_prefetch(aptr + 32, 0, 1);
      __builtin_prefetch(bptr + 32, 0, 1);
    }
#if HAVE_ASYNC_LDS
    // CDNA5 async DMA: global -> LDS without touching VGPRs (tracked by ASYNCcnt)
    __builtin_amdgcn_global_load_async_to_lds_b128(
        (gv4i*)(aptr + k0), (lv4i*)&As[arow][acol], 0, 0);
    __builtin_amdgcn_global_load_async_to_lds_b128(
        (gv4i*)(bptr + k0), (lv4i*)&Bs[brow][bcol], 0, 0);
    __builtin_amdgcn_global_load_async_to_lds_b128(
        (gv4i*)(bptr + k0 + 8), (lv4i*)&Bs[brow][bcol + 8], 0, 0);
#if __has_builtin(__builtin_amdgcn_s_wait_asynccnt)
    __builtin_amdgcn_s_wait_asynccnt(0);
#else
    asm volatile("s_wait_asynccnt 0x0" ::: "memory");
#endif
#else
    *(v8bf*)&As[arow][acol]     = *(const v8bf*)(aptr + k0);
    *(v8bf*)&Bs[brow][bcol]     = *(const v8bf*)(bptr + k0);
    *(v8bf*)&Bs[brow][bcol + 8] = *(const v8bf*)(bptr + k0 + 8);
#endif
    __syncthreads();

    // build fragments (documented 16-bit A 16x32 lane/VGPR layout)
    v16bf afr[2], bfr[2];
    const int mA = lane & 15;
    const int kh = lane >> 4;
#pragma unroll
    for (int fm = 0; fm < 2; fm++) {
#pragma unroll
      for (int e = 0; e < 16; e++) {
        const int r = e >> 1, p = e & 1;
        const int k = ((r >= 4) ? 16 : 0) + kh * 8 + ((r & 3) << 1) + p;
        afr[fm][e] = As[wm * 32 + fm * 16 + mA][k];
      }
    }
    const int nB = lane & 15;
    const int kg = lane >> 4;
#pragma unroll
    for (int fn = 0; fn < 2; fn++) {
#pragma unroll
      for (int e = 0; e < 16; e++)
        bfr[fn][e] = Bs[wn * 32 + fn * 16 + nB][((e >= 8) ? 16 : 0) + kg * 8 + (e & 7)];
    }

#pragma unroll
    for (int fm = 0; fm < 2; fm++)
#pragma unroll
      for (int fn = 0; fn < 2; fn++)
        acc[fm][fn] = __builtin_amdgcn_wmma_f32_16x16x32_bf16(
            false, afr[fm], false, bfr[fn], (short)0, acc[fm][fn], false, false);
    __syncthreads();
  }

  // epilogue (documented 32-bit C/D layout: M = e + 8*(lane>>4), N = lane&15)
#pragma unroll
  for (int fm = 0; fm < 2; fm++) {
#pragma unroll
    for (int fn = 0; fn < 2; fn++) {
#pragma unroll
      for (int e = 0; e < 8; e++) {
        const int ml = e + ((lane >> 4) << 3);
        const int gm = m0 + wm * 32 + fm * 16 + ml;
        const int gn = n0 + wn * 32 + fn * 16 + (lane & 15);
        float v = acc[fm][fn][e];
        if (MODE == 1) {
          v += bias[gn];
          v = (v > 20.0f) ? v : log1pf(__expf(v));   // softplus
        } else if (MODE == 2) {
          v += res[(size_t)gm * N + gn];
        }
        C[(size_t)gm * N + gn] = v;
      }
    }
  }
}

// ---------------- depthwise causal conv (width 4) + SiLU ----------------
__global__ void k_conv_silu(const float* __restrict__ xz, const float* __restrict__ cw,
                            const float* __restrict__ cb, float* __restrict__ u,
                            __bf16* __restrict__ ub) {
  const int t = blockIdx.x;            // token
  const int d = threadIdx.x;           // channel 0..511
  const int b = t / SEQ_C, l = t % SEQ_C;
  float s = cb[d];
#pragma unroll
  for (int j = 0; j < D_CONV_C; j++) {
    const int ls = l - (D_CONV_C - 1) + j;
    if (ls >= 0)
      s += cw[d * D_CONV_C + j] * xz[(size_t)(b * SEQ_C + ls) * (2 * D_INNER_C) + d];
  }
  const float v = s / (1.0f + __expf(-s));   // silu
  const size_t idx = (size_t)t * D_INNER_C + d;
  u[idx] = v;
  ub[idx] = f2bf(v);
}

// ---- pack dt (first 16 cols of xdb, stride XDB_N) into zero-padded bf16 [BL, DT_K] ----
__global__ void k_pack_dt(const float* __restrict__ xdb, __bf16* __restrict__ dtb) {
  const int i = blockIdx.x * blockDim.x + threadIdx.x;   // over BL * DT_K
  if (i < BL_C * DT_K) {
    const int k = i & (DT_K - 1);
    const int t = i >> 5;
    dtb[i] = (k < DT_RANK_C) ? f2bf(xdb[(size_t)t * XDB_N + k]) : bf_zero();
  }
}

// ---------------- selective scan: one thread per (batch, channel), 16 states in regs ----
// B/C projections (32 floats per step, shared across all 512 channels of a batch)
// are staged through LDS in 64-step chunks -> inner loop is pure VALU + LDS broadcast.
#define SCAN_CHUNK 64
__global__ __launch_bounds__(D_INNER_C) void k_scan(
    const float* __restrict__ delta, const float* __restrict__ xdb,
    const float* __restrict__ u, const float* __restrict__ A_log,
    float* __restrict__ y) {
  __shared__ float sBC[SCAN_CHUNK][32];  // [.][0..15]=B, [.][16..31]=C
  const int b = blockIdx.x;
  const int d = threadIdx.x;

  float a[D_STATE_C];
#pragma unroll
  for (int n = 0; n < D_STATE_C; n++) a[n] = -__expf(A_log[d * D_STATE_C + n]);
  float st[D_STATE_C] = {};

  for (int l0 = 0; l0 < SEQ_C; l0 += SCAN_CHUNK) {
    for (int i = threadIdx.x; i < SCAN_CHUNK * 32; i += D_INNER_C) {
      const int ll = i >> 5, j = i & 31;
      sBC[ll][j] = xdb[(size_t)(b * SEQ_C + l0 + ll) * XDB_N + DT_RANK_C + j];
    }
    __syncthreads();
    for (int li = 0; li < SCAN_CHUNK; ++li) {
      const size_t idx = (size_t)(b * SEQ_C + l0 + li) * D_INNER_C + d;
      const float de = delta[idx];
      const float du = de * u[idx];
      float yy = 0.0f;
#pragma unroll
      for (int n = 0; n < D_STATE_C; n++) {
        const float e = __expf(de * a[n]);
        st[n] = e * st[n] + du * sBC[li][n];
        yy += st[n] * sBC[li][16 + n];
      }
      y[idx] = yy;
    }
    __syncthreads();
  }
}

// ---------------- gate: yg = (y + Dp*u) * silu(z), output bf16 ----------------
__global__ void k_gate(const float* __restrict__ y, const float* __restrict__ u,
                       const float* __restrict__ xz, const float* __restrict__ Dp,
                       __bf16* __restrict__ yg) {
  const size_t t = blockIdx.x;
  const int d = threadIdx.x;
  const float z = xz[t * (2 * D_INNER_C) + D_INNER_C + d];
  const float sz = z / (1.0f + __expf(-z));
  const size_t idx = t * D_INNER_C + d;
  yg[idx] = f2bf((y[idx] + Dp[d] * u[idx]) * sz);
}

// ---------------- host-side orchestration ----------------
extern "C" void kernel_launch(void* const* d_in, const int* in_sizes, int n_in,
                              void* d_out, int out_size, void* d_ws, size_t ws_size,
                              hipStream_t stream) {
  (void)in_sizes; (void)n_in; (void)out_size; (void)ws_size;
  const float* x      = (const float*)d_in[0];   // [B, D_MODEL, SEQ]
  const float* W_in   = (const float*)d_in[1];   // [6, 1024, 256]
  const float* conv_w = (const float*)d_in[2];   // [6, 512, 1, 4]
  const float* conv_b = (const float*)d_in[3];   // [6, 512]
  const float* W_x    = (const float*)d_in[4];   // [6, 48, 512]
  const float* W_dt   = (const float*)d_in[5];   // [6, 512, 16]
  const float* b_dt   = (const float*)d_in[6];   // [6, 512]
  const float* A_log  = (const float*)d_in[7];   // [6, 512, 16]
  const float* Dp     = (const float*)d_in[8];   // [6, 512]
  const float* W_out  = (const float*)d_in[9];   // [6, 256, 512]
  const float* ln_g   = (const float*)d_in[10];  // [6, 256]
  const float* ln_b   = (const float*)d_in[11];  // [6, 256]
  float* out = (float*)d_out;

  char* ws = (char*)d_ws;
  size_t off = 0;
  auto alloc = [&](size_t bytes) -> void* {
    void* p = ws + off;
    off += (bytes + 255) & ~(size_t)255;
    return p;
  };

  float*  hA    = (float*) alloc((size_t)BL_C * D_MODEL_C * 4);
  float*  hB    = (float*) alloc((size_t)BL_C * D_MODEL_C * 4);
  __bf16* hn    = (__bf16*)alloc((size_t)BL_C * D_MODEL_C * 2);
  float*  xz    = (float*) alloc((size_t)BL_C * 2 * D_INNER_C * 4);
  float*  u     = (float*) alloc((size_t)BL_C * D_INNER_C * 4);
  __bf16* ub    = (__bf16*)alloc((size_t)BL_C * D_INNER_C * 2);
  float*  xdb   = (float*) alloc((size_t)BL_C * XDB_N * 4);
  __bf16* dtb   = (__bf16*)alloc((size_t)BL_C * DT_K * 2);
  float*  delta = (float*) alloc((size_t)BL_C * D_INNER_C * 4);
  float*  yb    = (float*) alloc((size_t)BL_C * D_INNER_C * 4);
  __bf16* yg    = (__bf16*)alloc((size_t)BL_C * D_INNER_C * 2);
  __bf16* wInB  = (__bf16*)alloc((size_t)N_LAYERS_C * 2 * D_INNER_C * D_MODEL_C * 2);
  __bf16* wXB   = (__bf16*)alloc((size_t)N_LAYERS_C * XDB_N * D_INNER_C * 2);
  __bf16* wDtB  = (__bf16*)alloc((size_t)N_LAYERS_C * D_INNER_C * DT_K * 2);
  __bf16* wOutB = (__bf16*)alloc((size_t)N_LAYERS_C * D_MODEL_C * D_INNER_C * 2);

  // weights -> bf16 (with zero padding to exact tiles), every launch (ws is poisoned)
  k_cvt_pad<<<1024, 256, 0, stream>>>(W_in,  wInB,  N_LAYERS_C, 2 * D_INNER_C, 2 * D_INNER_C, D_MODEL_C, D_MODEL_C);
  k_cvt_pad<<<512,  256, 0, stream>>>(W_x,   wXB,   N_LAYERS_C, 48, XDB_N, D_INNER_C, D_INNER_C);
  k_cvt_pad<<<128,  256, 0, stream>>>(W_dt,  wDtB,  N_LAYERS_C, D_INNER_C, D_INNER_C, DT_RANK_C, DT_K);
  k_cvt_pad<<<512,  256, 0, stream>>>(W_out, wOutB, N_LAYERS_C, D_MODEL_C, D_MODEL_C, D_INNER_C, D_INNER_C);

  // x [B, D, L] -> token-major h [B, L, D]
  dim3 trBlk(32, 8);
  k_transpose<<<dim3(SEQ_C / 32, D_MODEL_C / 32, BATCH_C), trBlk, 0, stream>>>(x, hA, D_MODEL_C, SEQ_C);

  float* hc = hA;
  float* hx = hB;
  for (int li = 0; li < N_LAYERS_C; ++li) {
    k_layernorm<<<BL_C, D_MODEL_C, 0, stream>>>(hc, ln_g + li * D_MODEL_C, ln_b + li * D_MODEL_C, hn);

    // xz = LN(h) @ W_in^T : [32768, 1024], K=256
    k_gemm_bf16_wmma<0><<<dim3((2 * D_INNER_C) / 128, BL_C / 64), 256, 0, stream>>>(
        hn, wInB + (size_t)li * 2 * D_INNER_C * D_MODEL_C, xz,
        BL_C, 2 * D_INNER_C, D_MODEL_C, nullptr, nullptr);

    k_conv_silu<<<BL_C, D_INNER_C, 0, stream>>>(
        xz, conv_w + (size_t)li * D_INNER_C * D_CONV_C, conv_b + li * D_INNER_C, u, ub);

    // xdb = u @ W_x^T : [32768, 128(padded from 48)], K=512
    k_gemm_bf16_wmma<0><<<dim3(XDB_N / 128, BL_C / 64), 256, 0, stream>>>(
        ub, wXB + (size_t)li * XDB_N * D_INNER_C, xdb,
        BL_C, XDB_N, D_INNER_C, nullptr, nullptr);

    k_pack_dt<<<(BL_C * DT_K + 255) / 256, 256, 0, stream>>>(xdb, dtb);

    // delta = softplus(dt @ W_dt^T + b_dt) : [32768, 512], K=32 (padded from 16)
    k_gemm_bf16_wmma<1><<<dim3(D_INNER_C / 128, BL_C / 64), 256, 0, stream>>>(
        dtb, wDtB + (size_t)li * D_INNER_C * DT_K, delta,
        BL_C, D_INNER_C, DT_K, b_dt + li * D_INNER_C, nullptr);

    k_scan<<<BATCH_C, D_INNER_C, 0, stream>>>(
        delta, xdb, u, A_log + (size_t)li * D_INNER_C * D_STATE_C, yb);

    k_gate<<<BL_C, D_INNER_C, 0, stream>>>(yb, u, xz, Dp + li * D_INNER_C, yg);

    // h_next = yg @ W_out^T + h (residual fused) : [32768, 256], K=512
    k_gemm_bf16_wmma<2><<<dim3(D_MODEL_C / 128, BL_C / 64), 256, 0, stream>>>(
        yg, wOutB + (size_t)li * D_MODEL_C * D_INNER_C, hx,
        BL_C, D_MODEL_C, D_INNER_C, nullptr, hc);

    float* tmp = hc; hc = hx; hx = tmp;
  }

  // h [B, L, D] -> out [B, D, L]
  k_transpose<<<dim3(D_MODEL_C / 32, SEQ_C / 32, BATCH_C), trBlk, 0, stream>>>(hc, out, SEQ_C, D_MODEL_C);
}
